// MultiHeadSelfAttention_19842748907702
// MI455X (gfx1250) — matrix operand order repaired and checked by reading, not compiled, not run on hardware
//
#include <hip/hip_runtime.h>
#include <hip/hip_bf16.h>
#include <math.h>

typedef __attribute__((ext_vector_type(16))) _Float16     v16h;
typedef __attribute__((ext_vector_type(8)))  float        v8f;
typedef __attribute__((ext_vector_type(4)))  unsigned int u32x4;

#define SEQ   2048
#define DM    1024
#define NH    16
#define DK    64
#define BATCH 2
#define MROWS (BATCH * SEQ)   // 4096

union Frag { u32x4 u[2]; v16h h; };

#if defined(__HIP_DEVICE_COMPILE__) && __has_builtin(__builtin_amdgcn_global_load_async_to_lds_b128)
#define HAVE_ASYNC_LDS 1
#endif

__device__ __forceinline__ v8f wmma16(v16h a, v16h b, v8f c) {
  // D = A(16x32 f16) x B(32x16 f16) + C(16x16 f32)
  return __builtin_amdgcn_wmma_f32_16x16x32_f16(false, a, false, b, (short)0, c,
                                                false, false);
}

// 16-byte global -> LDS copy; async (ASYNCcnt-tracked) when available.
#ifdef HAVE_ASYNC_LDS
typedef int i32x4v __attribute__((vector_size(16)));
typedef __attribute__((address_space(1))) i32x4v* g_b128_ptr;
typedef __attribute__((address_space(3))) i32x4v* l_b128_ptr;
#endif

__device__ __forceinline__ void copy16_to_lds(const _Float16* g, _Float16* l) {
#ifdef HAVE_ASYNC_LDS
  __builtin_amdgcn_global_load_async_to_lds_b128((g_b128_ptr)g, (l_b128_ptr)l,
                                                 0, 0);
#else
  *(u32x4*)l = *(const u32x4*)g;
#endif
}

__device__ __forceinline__ void wait_lds_copies() {
#ifdef HAVE_ASYNC_LDS
#if __has_builtin(__builtin_amdgcn_s_wait_asynccnt)
  __builtin_amdgcn_s_wait_asynccnt(0);
#else
  asm volatile("s_wait_asynccnt 0" ::: "memory");
#endif
  asm volatile("" ::: "memory");  // keep LDS reads below the wait
#endif
}

__global__ void cvt_f32_to_f16_kernel(const float* __restrict__ src,
                                      _Float16* __restrict__ dst, int n) {
  int i = blockIdx.x * blockDim.x + threadIdx.x;
  int stride = gridDim.x * blockDim.x;
  for (; i < n; i += stride) dst[i] = (_Float16)src[i];
}

// C[m,n] = sum_k A[m,k] * W[n,k]   (A: MROWSxDM f16, W: DMxDM f16 row-major)
// Each wave computes a 16(M) x 64(N) tile with 4 f32 accumulators.
// mode 0: scatter f16 to [B, H, S, DK] layout, optional fused RoPE (rope=1)
// mode 1: write f32 row-major [MROWS, DM] to outf
__global__ void gemm_xwT_kernel(const _Float16* __restrict__ A,
                                const _Float16* __restrict__ W,
                                _Float16* __restrict__ outh,
                                float* __restrict__ outf,
                                const int* __restrict__ pos,
                                int mode, int rope) {
  const int lane = threadIdx.x & 31;
  const int w    = blockIdx.x * (blockDim.x >> 5) + (threadIdx.x >> 5);
  const int mt   = w >> 4;   // 256 m-tiles of 16
  const int nt   = w & 15;   // 16 n-tiles of 64
  if (mt >= (MROWS >> 4)) return;
  const int m0 = mt << 4, n0 = nt << 6;
  const int l15 = lane & 15;
  const bool hiL = lane >= 16;

  v8f acc[4] = {};

  // A-frag: lane m = l15; K chunks {base..base+7, base+16..base+23}, base=hi?8:0
  const _Float16* arow = A + (size_t)(m0 + l15) * DM + (hiL ? 8 : 0);
  // B-frag: lane n = l15; K chunks {base..base+7, base+16..base+23}, base=hi?8:0
  const int koffB = hiL ? 8 : 0;
  const _Float16* wbase0 = W + (size_t)(n0 + l15) * DM + koffB;

  for (int k = 0; k < DM; k += 32) {
    // speculative prefetch two k-steps ahead (global_prefetch_b8; faults dropped)
    __builtin_prefetch(arow + k + 64, 0, 3);
    __builtin_prefetch(wbase0 + k + 64, 0, 3);
    Frag a;
    a.u[0] = *(const u32x4*)(arow + k);
    a.u[1] = *(const u32x4*)(arow + k + 16);
#pragma unroll
    for (int f = 0; f < 4; ++f) {
      const _Float16* wrow = wbase0 + (size_t)(f * 16) * DM + k;
      Frag bf;
      bf.u[0] = *(const u32x4*)(wrow);
      bf.u[1] = *(const u32x4*)(wrow + 16);
      acc[f] = wmma16(a.h, bf.h, acc[f]);
    }
  }

  if (mode == 1) {
#pragma unroll
    for (int r = 0; r < 8; ++r) {
      const int m = m0 + r + (hiL ? 8 : 0);
#pragma unroll
      for (int f = 0; f < 4; ++f) {
        const int n = n0 + f * 16 + l15;
        outf[(size_t)m * DM + n] = acc[f][r];
      }
    }
    return;
  }

  // mode 0: scatter f16 to [B,H,S,DK], with optional fused RoPE.
  float freqv[4]; int dloc[4]; int hh[4];
#pragma unroll
  for (int f = 0; f < 4; ++f) {
    const int n = n0 + f * 16 + l15;
    hh[f]   = n >> 6;
    dloc[f] = n & (DK - 1);
    const int p = dloc[f] >> 1;                       // pair index
    freqv[f] = __powf(10000.0f, -(float)(2 * p) * (1.0f / 64.0f));
  }
  const int odd = l15 & 1;
#pragma unroll
  for (int r = 0; r < 8; ++r) {
    const int m = m0 + r + (hiL ? 8 : 0);
    const int b = m >> 11, s = m & (SEQ - 1);
    const float pp = rope ? (float)pos[s] : 0.0f;
#pragma unroll
    for (int f = 0; f < 4; ++f) {
      float v = acc[f][r];
      float partner = __shfl_xor(v, 1, 32);           // (even,odd) pair mate
      float res = v;
      if (rope) {
        float sn, cs;
        __sincosf(pp * freqv[f], &sn, &cs);
        res = odd ? (partner * sn + v * cs)           // r2 = x1*sin + x2*cos
                  : (v * cs - partner * sn);          // r1 = x1*cos - x2*sin
      }
      const size_t idx = (((size_t)(b * NH + hh[f]) * SEQ + s) * DK) + dloc[f];
      outh[idx] = (_Float16)res;
    }
  }
}

// Flash-style causal attention. One wave per (b, head, 16-query tile).
// Computes S^T = K_tile x Q^T so the exp'd scores land lane-aligned with the
// A-fragment needed for P x V (no LDS transpose needed for P).
// The 32x64 V tile is staged into a per-wave LDS slice with 128-bit
// async copies (ASYNCcnt) overlapping the S^T WMMAs + softmax VALU work;
// the column-major B-frag gather then hits LDS instead of VMEM.
__global__ void attn_kernel(const _Float16* __restrict__ Qh,
                            const _Float16* __restrict__ Kh,
                            const _Float16* __restrict__ Vh,
                            _Float16* __restrict__ Oh) {
  __shared__ _Float16 vt[8][32][DK];    // 8 waves x 4KB = 32KB
  const int lane = threadIdx.x & 31;
  const int wv   = threadIdx.x >> 5;
  const int w    = blockIdx.x * (blockDim.x >> 5) + wv;
  const int qt   = w & 127;
  const int head = (w >> 7) & 15;
  const int b    = w >> 11;
  if (b >= BATCH) return;
  const int q0  = qt << 4;
  const int l15 = lane & 15;
  const bool hiL = lane >= 16;
  const size_t bh = (size_t)(b * NH + head) * SEQ * DK;
  const _Float16* Qb = Qh + bh;
  const _Float16* Kb = Kh + bh;
  const _Float16* Vb = Vh + bh;

  // Q^T B-frags: lane n = query; frag0 = dims 0..31, frag1 = dims 32..63
  Frag bq0, bq1;
  {
    const _Float16* qrow = Qb + (size_t)(q0 + l15) * DK + (hiL ? 8 : 0);
    bq0.u[0] = *(const u32x4*)(qrow);
    bq0.u[1] = *(const u32x4*)(qrow + 16);
    bq1.u[0] = *(const u32x4*)(qrow + 32);
    bq1.u[1] = *(const u32x4*)(qrow + 48);
  }

  float run_m = -1e30f, run_l = 0.0f;
  v8f acc[4] = {};                     // 16 q x 64 d output accumulator
  const float scale = 0.125f;          // 1/sqrt(64)
  const int ab  = hiL ? 8 : 0;
  const int nkb = (q0 + 16 + 31) >> 5; // causal: key blocks of 32

  for (int kb = 0; kb < nkb; ++kb) {
    const int k0 = kb << 5;

    // Kick off V-tile staging (lane L copies key row k0+L, 128 bytes).
    {
      const _Float16* gsrc = Vb + (size_t)(k0 + lane) * DK;
      _Float16* ldst = &vt[wv][lane][0];
#pragma unroll
      for (int i = 0; i < 8; ++i) copy16_to_lds(gsrc + i * 8, ldst + i * 8);
    }
    // Prefetch next block's K rows while we compute.
    __builtin_prefetch(Kb + (size_t)(k0 + 32 + l15) * DK, 0, 3);
    __builtin_prefetch(Kb + (size_t)(k0 + 48 + l15) * DK, 0, 3);

    float sc[2][8];
#pragma unroll
    for (int t = 0; t < 2; ++t) {
      // A-frag = K rows (16 keys x 64 dims, two k-steps of 32)
      const _Float16* krow = Kb + (size_t)(k0 + t * 16 + l15) * DK + ab;
      Frag a0, a1;
      a0.u[0] = *(const u32x4*)(krow);
      a0.u[1] = *(const u32x4*)(krow + 16);
      a1.u[0] = *(const u32x4*)(krow + 32);
      a1.u[1] = *(const u32x4*)(krow + 48);
      v8f c = {};
      c = wmma16(a0.h, bq0.h, c);
      c = wmma16(a1.h, bq1.h, c);
#pragma unroll
      for (int r = 0; r < 8; ++r) sc[t][r] = c[r];
    }
    // scale + causal mask; S^T layout: lane = query(l15), vgpr r = key
    float tmax = -1e30f;
#pragma unroll
    for (int t = 0; t < 2; ++t)
#pragma unroll
      for (int r = 0; r < 8; ++r) {
        const int gk = k0 + t * 16 + r + (hiL ? 8 : 0);
        float v = sc[t][r] * scale;
        if (gk > q0 + l15) v = -1e30f;
        sc[t][r] = v;
        tmax = fmaxf(tmax, v);
      }
    tmax = fmaxf(tmax, __shfl_xor(tmax, 16, 32));
    const float new_m = fmaxf(run_m, tmax);
    const float alpha = __expf(run_m - new_m);
    float lsum = 0.0f;
    Frag pa;                           // P A-frag: lane-aligned by construction
#pragma unroll
    for (int t = 0; t < 2; ++t)
#pragma unroll
      for (int r = 0; r < 8; ++r) {
        const float pv = __expf(sc[t][r] - new_m);
        lsum += pv;
        pa.h[t * 8 + r] = (_Float16)pv;
      }
    lsum += __shfl_xor(lsum, 16, 32);
    run_l = run_l * alpha + lsum;
    run_m = new_m;
    // rescale accumulator: broadcast per-query alpha into the D layout
    float am[8];
#pragma unroll
    for (int r = 0; r < 8; ++r) am[r] = __shfl(alpha, r + (hiL ? 8 : 0), 32);
#pragma unroll
    for (int f = 0; f < 4; ++f)
#pragma unroll
      for (int r = 0; r < 8; ++r) acc[f][r] *= am[r];

    // V B-frags from LDS: lane n = dim, keys {base..base+7, base+16..base+23}, base=hi?8:0
    wait_lds_copies();
    const int krl = hiL ? 8 : 0;
#pragma unroll
    for (int f = 0; f < 4; ++f) {
      const int d = f * 16 + l15;
      Frag bv;
#pragma unroll
      for (int j = 0; j < 16; ++j) bv.h[j] = vt[wv][krl + j + (j & 8)][d];
      acc[f] = wmma16(pa.h, bv.h, acc[f]);
    }
  }

  float lm[8];
#pragma unroll
  for (int r = 0; r < 8; ++r) lm[r] = __shfl(run_l, r + (hiL ? 8 : 0), 32);
#pragma unroll
  for (int r = 0; r < 8; ++r) {
    const int s = q0 + r + (hiL ? 8 : 0);
    const float inv = 1.0f / lm[r];
#pragma unroll
    for (int f = 0; f < 4; ++f) {
      const int d = f * 16 + l15;
      // layout [B, S, H, DK] == rows of the final projection GEMM
      Oh[(((size_t)b * SEQ + s) * NH + head) * DK + d] = (_Float16)(acc[f][r] * inv);
    }
  }
}

extern "C" void kernel_launch(void* const* d_in, const int* in_sizes, int n_in,
                              void* d_out, int out_size, void* d_ws, size_t ws_size,
                              hipStream_t stream) {
  (void)in_sizes; (void)n_in; (void)out_size; (void)ws_size;
  const float* x   = (const float*)d_in[0];
  const int*   pos = (const int*)d_in[1];
  const float* wq  = (const float*)d_in[2];
  const float* wk  = (const float*)d_in[3];
  const float* wv  = (const float*)d_in[4];
  const float* wo  = (const float*)d_in[5];
  float* out = (float*)d_out;

  char* ws = (char*)d_ws;
  const size_t MB = 1024 * 1024;
  _Float16* xh  = (_Float16*)(ws + 0);        //  8 MB: x f16 [4096,1024]
  _Float16* wqh = (_Float16*)(ws + 8 * MB);   //  2 MB each
  _Float16* wkh = (_Float16*)(ws + 10 * MB);
  _Float16* wvh = (_Float16*)(ws + 12 * MB);
  _Float16* woh = (_Float16*)(ws + 14 * MB);
  _Float16* Qh  = (_Float16*)(ws + 16 * MB);  //  8 MB [B,H,S,DK]
  _Float16* Kh  = (_Float16*)(ws + 24 * MB);
  _Float16* Vh  = (_Float16*)(ws + 32 * MB);
  _Float16* Oh  = (_Float16*)(ws + 40 * MB);  //  8 MB [B,S,H*DK]; total 48 MB

  cvt_f32_to_f16_kernel<<<2048, 256, 0, stream>>>(x,  xh,  MROWS * DM);
  cvt_f32_to_f16_kernel<<<1024, 256, 0, stream>>>(wq, wqh, DM * DM);
  cvt_f32_to_f16_kernel<<<1024, 256, 0, stream>>>(wk, wkh, DM * DM);
  cvt_f32_to_f16_kernel<<<1024, 256, 0, stream>>>(wv, wvh, DM * DM);
  cvt_f32_to_f16_kernel<<<1024, 256, 0, stream>>>(wo, woh, DM * DM);

  const int gemm_blocks = (MROWS / 16) * (DM / 64) / 8;  // 512 blocks x 8 waves
  gemm_xwT_kernel<<<gemm_blocks, 256, 0, stream>>>(xh, wqh, Qh, nullptr, pos, 0, 1);
  gemm_xwT_kernel<<<gemm_blocks, 256, 0, stream>>>(xh, wkh, Kh, nullptr, pos, 0, 1);
  gemm_xwT_kernel<<<gemm_blocks, 256, 0, stream>>>(xh, wvh, Vh, nullptr, pos, 0, 0);

  const int attn_blocks = (BATCH * NH * (SEQ / 16)) / 8; // 512
  attn_kernel<<<attn_blocks, 256, 0, stream>>>(Qh, Kh, Vh, Oh);

  gemm_xwT_kernel<<<gemm_blocks, 256, 0, stream>>>(Oh, woh, nullptr, out, nullptr, 1, 0);
}
